// WNO2d_17978733101703
// MI455X (gfx1250) — compile-verified, matrix operand order checked by reading, not run on hardware
//
#include <hip/hip_runtime.h>
#include <hip/hip_bf16.h>

// ---------------------------------------------------------------------------
// WNO2d forward for MI455X (gfx1250, wave32, WMMA).
// Activations NHWC [b][y][x][c] (c=35 innermost); wno_w pre-transposed per
// subband so the 96MB weight stream is coalesced (HBM roofline, 23.3 TB/s).
// GEMMs use v_wmma_f32_16x16x32_bf16. Operands are staged in LDS as
// ZERO-PADDED bf16 in fragment layout: each lane's 16 elements are two
// contiguous 16B runs -> unconditional ds_load_b128, no guards/converts in
// the inner WMMA loops.
// ---------------------------------------------------------------------------

typedef __attribute__((ext_vector_type(16))) __bf16 v16bf;
typedef __attribute__((ext_vector_type(8)))  __bf16 v8bf;
typedef __attribute__((ext_vector_type(8)))  float  v8f;

#define NB   32      // batch
#define NC   35      // channels (W)
#define NS   64      // spatial
#define NM   35      // wavelet spatial
#define NSTP 1225    // NM*NM
#define NPIX 4096    // NS*NS
#define KP   64      // padded K (35 -> 64)
#define NP   48      // padded N (35 -> 48)

__constant__ float c_dlo[8] = {
  -0.010597401784997278f,  0.032883011666982945f,  0.030841381835986965f,
  -0.18703481171888114f,  -0.02798376941698385f,   0.6308807679295904f,
   0.7148465705525415f,    0.23037781330885523f };

__device__ __forceinline__ float dlo(int k){ return c_dlo[k]; }
// DEC_HI[k] = (-1)^(k+1) * DEC_LO[7-k]
__device__ __forceinline__ float dhi(int k){ return ((k & 1) ? 1.0f : -1.0f) * c_dlo[7 - k]; }
__device__ __forceinline__ int refl64(int u){ u = (u < 0) ? (-u - 1) : u; return (u >= NS) ? (2 * NS - 1 - u) : u; }

// Fragment load from a zero-padded bf16 row of KP elements.
// Lane pattern (ISA 7.12.2): j<8 -> k = k0 + 8*half + j ; j>=8 -> +16.
// Both runs are 16B-aligned -> two ds_load_b128.
__device__ __forceinline__ v16bf frag_ld(const __bf16* row, int k0) {
  int half = (threadIdx.x & 31) >> 4;
  const __bf16* q = row + k0 + 8 * half;
  v8bf lo = *(const v8bf*)(q);
  v8bf hi = *(const v8bf*)(q + 16);
  v16bf v;
#pragma unroll
  for (int j = 0; j < 8; ++j) { v[j] = lo[j]; v[j + 8] = hi[j]; }
  return v;
}

#define WMMA_BF16(A, B, C) \
  __builtin_amdgcn_wmma_f32_16x16x32_bf16(false, (A), false, (B), (short)0, (C), false, false)

// ---------------------------------------------------------------------------
// fc0: h[b][y][x][c] = concat(x, grid) @ fc0_w + fc0_b   (K=5, VALU kernel)
// ---------------------------------------------------------------------------
__global__ void fc0_kernel(const float* __restrict__ x, const float* __restrict__ w,
                           const float* __restrict__ b, float* __restrict__ h) {
  int idx = blockIdx.x * blockDim.x + threadIdx.x;
  if (idx >= NB * NPIX * NC) return;
  int c = idx % NC; int p = idx / NC;
  int j = p % NS; int t = p / NS;
  int i = t % NS; int bb = t / NS;
  const float* xr = x + (((size_t)bb * NS + i) * NS + j) * 3;
  float gx = i * (1.0f / 63.0f), gy = j * (1.0f / 63.0f);
  float acc = b[c];
  acc += xr[0] * w[0 * NC + c] + xr[1] * w[1 * NC + c] + xr[2] * w[2 * NC + c]
       + gx * w[3 * NC + c] + gy * w[4 * NC + c];
  h[idx] = acc;
}

// ---------------------------------------------------------------------------
// DWT row pass: out[f][b][y][t][c], symmetric pad, stride 2, 8 taps
// ---------------------------------------------------------------------------
__global__ void dwtx_kernel(const float* __restrict__ h, float* __restrict__ out) {
  int idx = blockIdx.x * blockDim.x + threadIdx.x;
  if (idx >= 2 * NB * NS * NM * NC) return;
  int c = idx % NC; int r = idx / NC;
  int t = r % NM; r /= NM;
  int y = r % NS; r /= NS;
  int b = r % NB; int f = r / NB;
  float acc = 0.0f;
#pragma unroll
  for (int k = 0; k < 8; ++k) {
    int u = refl64(2 * t + k - 6);
    float coef = f ? dhi(7 - k) : dlo(7 - k);
    acc += h[(((size_t)b * NS + y) * NS + u) * NC + c] * coef;
  }
  out[idx] = acc;
}

// DWT column pass: subs[s][b][ss][t][c], s = 2*f + g (0=ll,1=lh,2=hl,3=hh)
__global__ void dwty_kernel(const float* __restrict__ lohix, float* __restrict__ subs) {
  int idx = blockIdx.x * blockDim.x + threadIdx.x;
  if (idx >= 4 * NB * NM * NM * NC) return;
  int c = idx % NC; int r = idx / NC;
  int t = r % NM; r /= NM;
  int ss = r % NM; r /= NM;
  int b = r % NB; int s = r / NB;
  int f = s >> 1, g = s & 1;
  float acc = 0.0f;
#pragma unroll
  for (int k = 0; k < 8; ++k) {
    int u = refl64(2 * ss + k - 6);
    float coef = g ? dhi(7 - k) : dlo(7 - k);
    acc += lohix[((((size_t)f * NB + b) * NS + u) * NM + t) * NC + c] * coef;
  }
  subs[idx] = acc;
}

// ---------------------------------------------------------------------------
// Weight transpose: wt[st][io] = w[io][st]  (1225x1225, LDS tiled, coalesced)
// ---------------------------------------------------------------------------
__global__ void transp_kernel(const float* __restrict__ w, float* __restrict__ wt) {
  __shared__ float tile[32][33];
  int x0 = blockIdx.x * 32, y0 = blockIdx.y * 32;
  int tx = threadIdx.x;
  for (int ty = threadIdx.y; ty < 32; ty += 8) {
    int r = y0 + ty, cc = x0 + tx;
    if (r < NSTP && cc < NSTP) tile[ty][tx] = w[(size_t)r * NSTP + cc];
  }
  __syncthreads();
  for (int ty = threadIdx.y; ty < 32; ty += 8) {
    int r = x0 + ty, cc = y0 + tx;
    if (r < NSTP && cc < NSTP) wt[(size_t)r * NSTP + cc] = tile[tx][ty];
  }
}

// ---------------------------------------------------------------------------
// mul2d: per spatial point st, Out[32 x 35] = In[32 x 35] @ W_st[35 x 35]
// One wave per st, 4 waves per WG. A staged as [32][KP] bf16 (zero-padded);
// W staged transposed as [NP][KP] bf16. 12 wmma per wave, frag loads are
// unconditional ds_load_b128 pairs.
// ---------------------------------------------------------------------------
__global__ void __launch_bounds__(128) mul2d_kernel(const float* __restrict__ subs,
                                                    const float* __restrict__ wt,
                                                    float* __restrict__ out, int s) {
  __shared__ __align__(16) __bf16 aL[4][32][KP];
  __shared__ __align__(16) __bf16 bT[4][NP][KP];
  __shared__ float cfun[4][32][36];
  int lane = threadIdx.x & 31, wv = threadIdx.x >> 5;
  int idx = lane & 15, half = lane >> 4;
  int st = blockIdx.x * 4 + wv;
  bool valid = st < NSTP;
  int stc = valid ? st : NSTP - 1;

  // zero-fill B pad (LDS ops from one wave stay in order)
  for (int t = lane; t < NP * KP; t += 32) ((__bf16*)bT[wv])[t] = (__bf16)0.0f;
  // stage A: rows = batch, zero-padded K
  for (int t = lane; t < 32 * KP; t += 32) {
    int m = t >> 6, k = t & (KP - 1);
    float v = (k < NC) ? subs[((((size_t)s * NB + m) * NSTP) + stc) * NC + k] : 0.0f;
    aL[wv][m][k] = (__bf16)v;
  }
  // stage B transposed: wt tile is contiguous [io = i*35+o]
  const float* wsrc = wt + (size_t)stc * NSTP;
  for (int t = lane; t < NSTP; t += 32) {
    int i = t / NC, o = t % NC;
    bT[wv][o][i] = (__bf16)wsrc[t];
  }
  if (lane == 0 && st + 4 < NSTP) __builtin_prefetch(wt + (size_t)(st + 4) * NSTP, 0, 1);
  __syncthreads();

#pragma unroll
  for (int mt = 0; mt < 2; ++mt) {
    const __bf16* arow = aL[wv][mt * 16 + idx];
    v16bf a0 = frag_ld(arow, 0);
    v16bf a1 = frag_ld(arow, 32);
#pragma unroll
    for (int nt = 0; nt < 3; ++nt) {
      const __bf16* brow = bT[wv][nt * 16 + idx];
      v16bf b0 = frag_ld(brow, 0);
      v16bf b1 = frag_ld(brow, 32);
      v8f acc = {};
      acc = WMMA_BF16(a0, b0, acc);
      acc = WMMA_BF16(a1, b1, acc);
      int n = nt * 16 + idx;
      if (n < NC) {
#pragma unroll
        for (int r = 0; r < 8; ++r) cfun[wv][mt * 16 + r + 8 * half][n] = acc[r];
      }
    }
  }
  __syncthreads();
  if (valid) {
    for (int bb = 0; bb < 32; ++bb) {
      float* op = out + ((((size_t)s * NB + bb) * NSTP) + st) * NC;
      for (int c = lane; c < NC; c += 32) op[c] = cfun[wv][bb][c];
    }
  }
}

// ---------------------------------------------------------------------------
// 1x1 conv: x2[p][o] = h[p][i] @ cw[o][i] + cb[o]; B tile shared per WG.
// ---------------------------------------------------------------------------
__global__ void __launch_bounds__(256) conv1x1_kernel(const float* __restrict__ h,
                                                      const float* __restrict__ cw,
                                                      const float* __restrict__ cb,
                                                      float* __restrict__ x2) {
  __shared__ __align__(16) __bf16 bT[NP][KP];    // cw transposed, zero-padded
  __shared__ __align__(16) __bf16 aL[8][16][KP];
  __shared__ float cbs[NC];
  __shared__ float cfun[8][16][36];
  int lane = threadIdx.x & 31, wv = threadIdx.x >> 5;
  int idx = lane & 15, half = lane >> 4;
  size_t p0 = ((size_t)blockIdx.x * 8 + wv) * 16;

  for (int t = threadIdx.x; t < NP * KP; t += 256) ((__bf16*)bT)[t] = (__bf16)0.0f;
  __syncthreads();
  for (int t = threadIdx.x; t < NSTP; t += 256) {
    int o = t / NC, i = t % NC;               // cw[o][i]
    bT[o][i] = (__bf16)cw[t];
  }
  if (threadIdx.x < NC) cbs[threadIdx.x] = cb[threadIdx.x];
  // stage A rows for this wave (16 points), zero-padded K
  for (int t = lane; t < 16 * KP; t += 32) {
    int m = t >> 6, k = t & (KP - 1);
    float v = (k < NC) ? h[(p0 + m) * NC + k] : 0.0f;
    aL[wv][m][k] = (__bf16)v;
  }
  __syncthreads();

  const __bf16* arow = aL[wv][idx];
  v16bf a0 = frag_ld(arow, 0);
  v16bf a1 = frag_ld(arow, 32);
#pragma unroll
  for (int nt = 0; nt < 3; ++nt) {
    const __bf16* brow = bT[nt * 16 + idx];
    v16bf b0 = frag_ld(brow, 0);
    v16bf b1 = frag_ld(brow, 32);
    v8f acc = {};
    acc = WMMA_BF16(a0, b0, acc);
    acc = WMMA_BF16(a1, b1, acc);
    int n = nt * 16 + idx;
    if (n < NC) {
      float bias = cbs[n];
#pragma unroll
      for (int r = 0; r < 8; ++r) cfun[wv][r + 8 * half][n] = acc[r] + bias;
    }
  }
  __syncthreads();
  for (int m = 0; m < 16; ++m) {
    float* op = x2 + (p0 + m) * NC;
    for (int c = lane; c < NC; c += 32) op[c] = cfun[wv][m][c];
  }
}

// ---------------------------------------------------------------------------
// IDWT column pass: loY[f][b][y][t][c];  y[n] = sum_k gf[k]*x[(n+k-1)/2]
// ---------------------------------------------------------------------------
__global__ void idwty_kernel(const float* __restrict__ subm, float* __restrict__ loY) {
  int idx = blockIdx.x * blockDim.x + threadIdx.x;
  if (idx >= 2 * NB * NS * NM * NC) return;
  int c = idx % NC; int r = idx / NC;
  int t = r % NM; r /= NM;
  int y = r % NS; r /= NS;
  int b = r % NB; int f = r / NB;
  const float* s0 = subm + (((size_t)(2 * f + 0) * NB + b) * NSTP) * NC;
  const float* s1 = subm + (((size_t)(2 * f + 1) * NB + b) * NSTP) * NC;
  float acc = 0.0f;
#pragma unroll
  for (int k = 0; k < 8; ++k) {
    int m = y + k - 1;
    if (m >= 0 && !(m & 1)) {
      int ss = m >> 1;
      if (ss < NM) {
        size_t o = ((size_t)ss * NM + t) * NC + c;
        acc += dlo(k) * s0[o] + dhi(k) * s1[o];
      }
    }
  }
  loY[idx] = acc;
}

// IDWT row pass, fused with + x2 and ReLU -> h_out[b][y][x][c]
__global__ void idwtx_kernel(const float* __restrict__ loY, const float* __restrict__ x2,
                             float* __restrict__ hout) {
  int idx = blockIdx.x * blockDim.x + threadIdx.x;
  if (idx >= NB * NPIX * NC) return;
  int c = idx % NC; int r = idx / NC;
  int x = r % NS; r /= NS;
  int y = r % NS; int b = r / NS;
  const float* l0 = loY + (((size_t)0 * NB + b) * NS) * NM * NC;
  const float* l1 = loY + (((size_t)1 * NB + b) * NS) * NM * NC;
  float acc = 0.0f;
#pragma unroll
  for (int k = 0; k < 8; ++k) {
    int m = x + k - 1;
    if (m >= 0 && !(m & 1)) {
      int t = m >> 1;
      if (t < NM) {
        size_t o = ((size_t)y * NM + t) * NC + c;
        acc += dlo(k) * l0[o] + dhi(k) * l1[o];
      }
    }
  }
  acc += x2[idx];
  hout[idx] = acc > 0.0f ? acc : 0.0f;
}

// ---------------------------------------------------------------------------
// Fused fc1(35->256)+ReLU+fc2(256->1). fc1 weights staged once per WG as a
// [256][KP] bf16 LDS block (32KB); per-tile fc2 dot folded into per-lane
// partials, reduced with shfl_xor over 16-lane segments.
// ---------------------------------------------------------------------------
__global__ void __launch_bounds__(256) fc12_kernel(const float* __restrict__ h,
                                                   const float* __restrict__ w1,
                                                   const float* __restrict__ b1,
                                                   const float* __restrict__ w2,
                                                   const float* __restrict__ b2,
                                                   float* __restrict__ out) {
  __shared__ __align__(16) __bf16 bT[256][KP];   // fc1_w transposed, padded
  __shared__ __align__(16) __bf16 aL[8][16][KP];
  __shared__ float b1s[256];
  __shared__ float w2s[256];
  int lane = threadIdx.x & 31, wv = threadIdx.x >> 5;
  int idx = lane & 15, half = lane >> 4;
  size_t p0 = ((size_t)blockIdx.x * 8 + wv) * 16;

  for (int t = threadIdx.x; t < 256 * KP; t += 256) ((__bf16*)bT)[t] = (__bf16)0.0f;
  __syncthreads();
  for (int t = threadIdx.x; t < NC * 256; t += 256) {
    int k = t >> 8, n = t & 255;               // w1[k][n], coalesced read
    bT[n][k] = (__bf16)w1[t];
  }
  b1s[threadIdx.x] = b1[threadIdx.x];
  w2s[threadIdx.x] = w2[threadIdx.x];
  for (int t = lane; t < 16 * KP; t += 32) {
    int m = t >> 6, k = t & (KP - 1);
    float v = (k < NC) ? h[(p0 + m) * NC + k] : 0.0f;
    aL[wv][m][k] = (__bf16)v;
  }
  __syncthreads();

  const __bf16* arow = aL[wv][idx];
  v16bf a0 = frag_ld(arow, 0);
  v16bf a1 = frag_ld(arow, 32);

  float accf[8] = {0, 0, 0, 0, 0, 0, 0, 0};
#pragma unroll
  for (int nt = 0; nt < 16; ++nt) {
    int n = nt * 16 + idx;
    const __bf16* brow = bT[n];
    v16bf b0 = frag_ld(brow, 0);
    v16bf bb = frag_ld(brow, 32);
    v8f acc = {};
    acc = WMMA_BF16(a0, b0, acc);
    acc = WMMA_BF16(a1, bb, acc);
    float w2n = w2s[n], bn = b1s[n];
#pragma unroll
    for (int r = 0; r < 8; ++r) {
      float v = acc[r] + bn;
      v = v > 0.0f ? v : 0.0f;
      accf[r] += v * w2n;
    }
  }
#pragma unroll
  for (int r = 0; r < 8; ++r)
    for (int off = 8; off; off >>= 1) accf[r] += __shfl_xor(accf[r], off, 16);
  if (idx == 0) {
    float bb2 = b2[0];
#pragma unroll
    for (int r = 0; r < 8; ++r) out[p0 + r + 8 * half] = accf[r] + bb2;
  }
}

// ---------------------------------------------------------------------------
extern "C" void kernel_launch(void* const* d_in, const int* in_sizes, int n_in,
                              void* d_out, int out_size, void* d_ws, size_t ws_size,
                              hipStream_t stream) {
  const float* x   = (const float*)d_in[0];
  const float* f0w = (const float*)d_in[1];
  const float* f0b = (const float*)d_in[2];
  const float* wno = (const float*)d_in[3];
  const float* cw  = (const float*)d_in[4];
  const float* cb  = (const float*)d_in[5];
  const float* f1w = (const float*)d_in[6];
  const float* f1b = (const float*)d_in[7];
  const float* f2w = (const float*)d_in[8];
  const float* f2b = (const float*)d_in[9];
  float* out = (float*)d_out;
  float* ws  = (float*)d_ws;

  const size_t NH  = (size_t)NB * NPIX * NC;        // 4,587,520
  const size_t NT2 = (size_t)2 * NB * NS * NM * NC; // 5,017,600
  const size_t NSB = (size_t)4 * NB * NSTP * NC;    // 5,488,000

  float* hA   = ws;
  float* hB   = hA + NH;
  float* tmp2 = hB + NH;          // dwtx out / idwty out (phases don't overlap)
  float* subs = tmp2 + NT2;
  float* subm = subs + NSB;
  float* x2   = subm + NSB;
  float* wt   = x2 + NH;          // 1225x1225 transposed weight (per subband)

  fc0_kernel<<<(int)((NH + 255) / 256), 256, 0, stream>>>(x, f0w, f0b, hA);

  for (int blk = 0; blk < 4; ++blk) {
    float* hin  = (blk & 1) ? hB : hA;
    float* hout = (blk & 1) ? hA : hB;

    dwtx_kernel<<<(int)((NT2 + 255) / 256), 256, 0, stream>>>(hin, tmp2);
    dwty_kernel<<<(int)((NSB + 255) / 256), 256, 0, stream>>>(tmp2, subs);
    conv1x1_kernel<<<1024, 256, 0, stream>>>(hin, cw + (size_t)blk * NC * NC,
                                             cb + (size_t)blk * NC, x2);
    for (int s = 0; s < 4; ++s) {
      const float* wsub = wno + ((size_t)(blk * 4 + s)) * NC * NC * NSTP;
      transp_kernel<<<dim3(39, 39), dim3(32, 8), 0, stream>>>(wsub, wt);
      mul2d_kernel<<<(NSTP + 3) / 4, 128, 0, stream>>>(subs, wt, subm, s);
    }
    idwty_kernel<<<(int)((NT2 + 255) / 256), 256, 0, stream>>>(subm, tmp2);
    idwtx_kernel<<<(int)((NH + 255) / 256), 256, 0, stream>>>(tmp2, x2, hout);
  }

  fc12_kernel<<<1024, 256, 0, stream>>>(hA, f1w, f1b, f2w, f2b, out);
}
